// TorchBiphasicAxonMapSpatial_52991306498201
// MI455X (gfx1250) — compile-verified, matrix operand order checked by reading, not run on hardware
//
#include <hip/hip_runtime.h>
#include <hip/hip_bf16.h>

typedef __attribute__((ext_vector_type(16))) _Float16 v16h;
typedef __attribute__((ext_vector_type(8)))  float    v8f;

#define BB 2
#define EE 100
#define EP 128      // E padded to 4 K-blocks of 32
#define PP 9216
#define SS 128
#define NEG_BIG (-3.0e38f)

#if defined(__HIP_DEVICE_COMPILE__) && __has_builtin(__builtin_amdgcn_exp2f)
#define EXP2F(x) __builtin_amdgcn_exp2f(x)
#else
#define EXP2F(x) exp2f(x)
#endif

__global__ __launch_bounds__(256)
void biphasic_axonmap_kernel(const float* __restrict__ ep,    // (B,E,3)
                             const float* __restrict__ mp,    // (B,12)
                             const float* __restrict__ d2_el, // (P,S,E)
                             const float* __restrict__ axon,  // (P,S,3)
                             float* __restrict__ out)         // (B,P)
{
    __shared__ float c1s[BB][EP];   // -log2e / size_den      (0 in pad)
    __shared__ float c2s[BB][EP];   // +log2e / streak_den    (0 in pad)
    __shared__ float lfs[BB][EP];   // log2(F_bright)         (NEG_BIG if fb==0 or pad)
    __shared__ float red[BB][SS];

    const int tid = threadIdx.x;
    const int p   = blockIdx.x;

    // ---- per-(b,e) coefficients: 256 threads cover 2 x 128 entries ----
    {
        const int b = tid >> 7;
        const int e = tid & 127;
        float c1 = 0.0f, c2 = 0.0f, lf = NEG_BIG;
        if (e < EE) {
            const float freq = ep[(b * EE + e) * 3 + 0];
            const float amp  = ep[(b * EE + e) * 3 + 1];
            const float pdur = ep[(b * EE + e) * 3 + 2];
            const float* m = mp + b * 12;
            const float rho = m[0], axl = m[1];
            const float a0 = m[2], a1 = m[3], a2 = m[4], a3 = m[5];
            const float a5 = m[7], a6 = m[8], a7 = m[9], a8 = m[10], a9 = m[11];
            const float scaled = (a1 + a0 * pdur) * amp;
            float Fb = a2 * scaled + a3 * freq;
            Fb = (scaled > 0.25f) ? Fb : 0.0f;   // AMP_CUTOFF
            Fb = (amp > 0.0f) ? Fb : 0.0f;
            const float Fs   = fmaxf(a5 * scaled + a6, 100.0f / (rho * rho));
            const float Fstr = fmaxf(a9 - a7 * __powf(pdur, a8), 100.0f / (axl * axl));
            const float sden  = 2.0f * rho * rho * Fs;
            const float stden = axl * axl * Fstr;
            const float LOG2E = 1.4426950408889634f;
            c1 = -LOG2E / sden;
            c2 =  LOG2E / stden;
            lf = (Fb > 0.0f) ? __log2f(Fb) : NEG_BIG;  // fold F_bright into exponent
        }
        c1s[b][e] = c1;
        c2s[b][e] = c2;
        lfs[b][e] = lf;
    }
    __syncthreads();

    // ---- main: wave w handles s in [16w, 16w+16); lane&15 = s column ----
    const int lane  = tid & 31;
    const int wave  = tid >> 5;
    const int col   = lane & 15;   // N column of WMMA = s within wave tile
    const int khalf = lane >> 4;   // which K-half this lane supplies for B
    const int s     = wave * 16 + col;

    const size_t row = (size_t)p * SS + s;
    const float contrib = axon[row * 3 + 2];
    const float* d2row  = d2_el + row * EE;

    // A matrix is constant: row 0 = all ones, rows 1..15 = zero. Built once.
    // 16-bit A layout: lane m holds row m (lanes m and m+16 hold its two K-halves),
    // so exactly lanes with (lane&15)==0 carry row 0.
    v16h Am;
#pragma unroll
    for (int i = 0; i < 16; ++i)
        Am[i] = (col == 0) ? (_Float16)1.0f : (_Float16)0.0f;

    v8f acc0; v8f acc1;
#pragma unroll
    for (int i = 0; i < 8; ++i) { acc0[i] = 0.0f; acc1[i] = 0.0f; }

#pragma unroll
    for (int kb = 0; kb < 4; ++kb) {
        const int e0 = kb * 32 + khalf * 16;   // this lane's 16 contiguous e's (B layout)

        // stream d2 once per block; shared between the two batches
        float d2v[16];
#pragma unroll
        for (int j4 = 0; j4 < 4; ++j4) {
            const int be = e0 + j4 * 4;
            float4 v;
            if (be + 4 <= EE) {
                v = *(const float4*)(d2row + be);       // global_load_b128, 16B aligned
            } else {
                v = make_float4(0.0f, 0.0f, 0.0f, 0.0f); // pad; lfs pad zeroes B here
            }
            d2v[j4 * 4 + 0] = v.x; d2v[j4 * 4 + 1] = v.y;
            d2v[j4 * 4 + 2] = v.z; d2v[j4 * 4 + 3] = v.w;
        }

#pragma unroll
        for (int b = 0; b < 2; ++b) {
            // B matrix (K=32 x N=16, f16): lane holds its s-column for 16 e's.
            // B[e,s] = F_bright * exp term = exp2(d2*c1 + (contrib*c2 + log2 Fb))
            v16h Bm;
#pragma unroll
            for (int j = 0; j < 16; ++j) {
                const int e = e0 + j;
                const float t = fmaf(contrib, c2s[b][e], lfs[b][e]);
                const float x = fmaf(d2v[j], c1s[b][e], t);
                Bm[j] = (_Float16)EXP2F(x);   // v_exp_f32 (native exp2)
            }

            if (b == 0) {
                acc0 = __builtin_amdgcn_wmma_f32_16x16x32_f16(
                    false, Am, false, Bm, (short)0, acc0, false, false);
            } else {
                acc1 = __builtin_amdgcn_wmma_f32_16x16x32_f16(
                    false, Am, false, Bm, (short)0, acc1, false, false);
            }
        }
    }

    // D row 0 lives in VGPR0 of lanes 0..15 (M=0..7 half); column = s
    if (lane < 16) {
        red[0][wave * 16 + lane] = acc0[0];
        red[1][wave * 16 + lane] = acc1[0];
    }
    __syncthreads();

    // max over the 128 s values, both batches in parallel
    {
        const int b = tid >> 7;
        const int i = tid & 127;
#pragma unroll
        for (int off = 64; off > 0; off >>= 1) {
            if (i < off) red[b][i] = fmaxf(red[b][i], red[b][i + off]);
            __syncthreads();
        }
        if (i == 0) {
            const float v = red[b][0];
            out[(size_t)b * PP + p] = (v > 0.0f) ? v : 0.0f;  // THRESH_PERCEPT = 0
        }
    }
}

extern "C" void kernel_launch(void* const* d_in, const int* in_sizes, int n_in,
                              void* d_out, int out_size, void* d_ws, size_t ws_size,
                              hipStream_t stream) {
    const float* ep    = (const float*)d_in[0];  // elec_params  (2,100,3)
    const float* mp    = (const float*)d_in[1];  // model_params (2,12)
    const float* d2_el = (const float*)d_in[2];  // d2_el        (9216,128,100)
    const float* axon  = (const float*)d_in[3];  // axon_contrib (9216,128,3)
    float* out = (float*)d_out;                  // (2,9216) -> (2,96,96)

    biphasic_axonmap_kernel<<<PP, 256, 0, stream>>>(ep, mp, d2_el, axon, out);
}
